// AudioTowerStatic_22857815949419
// MI455X (gfx1250) — compile-verified, matrix operand order checked by reading, not run on hardware
//
#include <hip/hip_runtime.h>
#include <hip/hip_bf16.h>

// ---------------- CDNA5 WMMA types ----------------
typedef __attribute__((ext_vector_type(16))) __bf16 v16bf;
typedef __attribute__((ext_vector_type(8)))  __bf16 v8bf;
typedef __attribute__((ext_vector_type(8)))  float  v8f;

// ---------------- Problem constants ----------------
#define C_MEL   128
#define T_IN    3000
#define NCHUNK  30
#define BASE    100
#define LCHUNK  50
#define DMODEL  1280
#define NHEAD   20
#define DHEAD   64
#define DFF     5120
#define NLAYER  12
#define DOUTP   2048
#define NTOK    1500      // 30 * 50
#define MCONV1  3000      // 30 * 100
#define K1      384       // 3 * 128
#define K2      3840      // 3 * 1280
#define LN_EPS  1e-5f

// LDS strides (elements) padded for bank-conflict-free b128 fragment loads.
#define ASTR    40        // GEMM A tile row stride   (5 x v8bf)
#define BSTR    40        // GEMM B tile col stride   (5 x v8bf)
#define QSTR    72        // attention tile row stride (9 x v8bf)

__device__ __forceinline__ float gelu_erf(float x) {
  return 0.5f * x * (1.0f + erff(x * 0.70710678118654752f));
}

// Pack 8 f32 -> 8 bf16 (v_cvt_pk_bf16_f32 pairs).
__device__ __forceinline__ v8bf pack8(float4 a, float4 b) {
  v8bf p;
  p[0] = (__bf16)a.x; p[1] = (__bf16)a.y; p[2] = (__bf16)a.z; p[3] = (__bf16)a.w;
  p[4] = (__bf16)b.x; p[5] = (__bf16)b.y; p[6] = (__bf16)b.z; p[7] = (__bf16)b.w;
  return p;
}

// Two contiguous 8-element K-runs -> one 16-element WMMA fragment.
__device__ __forceinline__ v16bf frag(v8bf lo, v8bf hi) {
  union { struct { v8bf lo, hi; } s; v16bf v; } u;
  u.s.lo = lo; u.s.hi = hi;
  return u.v;
}

// =====================================================================
// GEMM: out[M,N] = epilogue(A[M,K] @ B[K,N]); A,B f32 in memory, bf16 in LDS.
// Block: 256 threads = 8 waves; 128x64 C tile; wave w owns rows w*16..+16,
// all 64 cols (4 accumulators). K % 32 == 0, N % 64 == 0.
// LDS: A [row][k] (K-contiguous), B transposed [col][k] (K-contiguous)
// so every fragment is two ds_load_b128.
// =====================================================================
__global__ __launch_bounds__(256)
void gemm_wmma_bf16(const float* __restrict__ A, const float* __restrict__ B,
                    const float* __restrict__ bias, const float* __restrict__ res,
                    int res_mod, float scale, int do_gelu,
                    float* __restrict__ out, int M, int N, int K) {
  __shared__ __bf16 As[128 * ASTR];   // 10 KB
  __shared__ __bf16 Bs[64 * BSTR];    //  5 KB

  const int tid  = threadIdx.x;
  const int lane = tid & 31;
  const int wave = tid >> 5;
  const int bm = blockIdx.y * 128;
  const int bn = blockIdx.x * 64;

  v8f acc[4] = {};

  for (int k0 = 0; k0 < K; k0 += 32) {
    if (k0 + 32 < K) {
      __builtin_prefetch(A + (long)(bm + (tid >> 1)) * K + (k0 + 32), 0, 0);
      __builtin_prefetch(B + (long)(k0 + 32 + (tid >> 3)) * N + bn + (tid & 7) * 8, 0, 0);
    }
    // ---- Stage A: 128 rows x 32 k; thread -> (row = tid/2, 16-elem half) ----
    {
      const int r = tid >> 1, half = tid & 1;
      const int gr = bm + r;
      v8bf p0 = {}, p1 = {};
      if (gr < M) {
        const float4* s4 = (const float4*)(A + (long)gr * K + k0 + half * 16);
        p0 = pack8(s4[0], s4[1]);
        p1 = pack8(s4[2], s4[3]);
      }
      v8bf* dst = (v8bf*)As + r * (ASTR / 8) + half * 2;
      dst[0] = p0; dst[1] = p1;
    }
    // ---- Stage B transposed: 32 k x 64 n; coalesced read, strided b16 store ----
    {
      const int r = tid >> 3, c0 = (tid & 7) * 8;
      const float4* s4 = (const float4*)(B + (long)(k0 + r) * N + bn + c0);
      v8bf p = pack8(s4[0], s4[1]);
#pragma unroll
      for (int j = 0; j < 8; ++j)
        Bs[(c0 + j) * BSTR + r] = p[j];
    }
    __syncthreads();

    // ---- Fragments: two v8bf (ds_load_b128) each ----
    const int kb8 = (lane & 16) ? 1 : 0;
    const v8bf* Av = (const v8bf*)As;
    const v8bf* Bv = (const v8bf*)Bs;
    const int arow = wave * 16 + (lane & 15);
    v16bf a = frag(Av[arow * (ASTR / 8) + kb8], Av[arow * (ASTR / 8) + kb8 + 2]);
#pragma unroll
    for (int nt = 0; nt < 4; ++nt) {
      const int col = nt * 16 + (lane & 15);
      v16bf b = frag(Bv[col * (BSTR / 8) + kb8], Bv[col * (BSTR / 8) + kb8 + 2]);
      acc[nt] = __builtin_amdgcn_wmma_f32_16x16x32_bf16(false, a, false, b, (short)0, acc[nt], false, false);
    }
    __syncthreads();
  }

  // ---- Epilogue ----
  const int rbase = bm + wave * 16 + ((lane & 16) ? 8 : 0);
  const int cb = bn + (lane & 15);
#pragma unroll
  for (int r = 0; r < 8; ++r) {
    const int row = rbase + r;
    if (row >= M) continue;
#pragma unroll
    for (int nt = 0; nt < 4; ++nt) {
      const int col = cb + nt * 16;
      float v = acc[nt][r];
      if (bias) v += bias[col];
      v *= scale;
      if (do_gelu) v = gelu_erf(v);
      if (res) {
        const int rr = res_mod ? (row % res_mod) : row;
        v += res[(long)rr * N + col];
      }
      out[(long)row * N + col] = v;
    }
  }
}

// =====================================================================
// Block-diagonal attention, one block per (chunk, head), 128 thr = 4 waves.
// Q pre-scaled by DH^-0.5. 50 rows padded to 64. LDS: Q,K row-major [t][dh],
// V transposed [dh][t] so all WMMA fragments are K-contiguous b128 loads.
// =====================================================================
__global__ __launch_bounds__(128)
void attn_wmma_kernel(const float* __restrict__ q, const float* __restrict__ k,
                      const float* __restrict__ v, float* __restrict__ o) {
  const int n  = blockIdx.x / NHEAD;
  const int hd = blockIdx.x % NHEAD;
  const int tid = threadIdx.x, lane = tid & 31, wave = tid >> 5;

  __shared__ __bf16 Qs[64 * QSTR];        // [row][dh]
  __shared__ __bf16 Ks[64 * QSTR];        // [key][dh]
  __shared__ __bf16 Vt[64 * QSTR];        // [dh][key]  (transposed)
  __shared__ float  Ss[4 * 16 * 64];
  __shared__ __bf16 Ps[4 * 16 * QSTR];

  // ---- Stage: 64x64 each; thread handles 8-element dh segments ----
#pragma unroll
  for (int it = 0; it < 4; ++it) {
    const int seg = tid + it * 128;        // 0..511
    const int row = seg >> 3, c0 = (seg & 7) * 8;
    v8bf zq = {}, zk = {}, zv = {};
    if (row < LCHUNK) {
      const long g = (long)(n * LCHUNK + row) * DMODEL + hd * DHEAD + c0;
      const float4* gq = (const float4*)(q + g);
      const float4* gk = (const float4*)(k + g);
      const float4* gv = (const float4*)(v + g);
      zq = pack8(gq[0], gq[1]);
      zk = pack8(gk[0], gk[1]);
      zv = pack8(gv[0], gv[1]);
    }
    ((v8bf*)Qs)[row * (QSTR / 8) + (seg & 7)] = zq;
    ((v8bf*)Ks)[row * (QSTR / 8) + (seg & 7)] = zk;
#pragma unroll
    for (int j = 0; j < 8; ++j)
      Vt[(c0 + j) * QSTR + row] = zv[j];
  }
  __syncthreads();

  const int kb8 = (lane & 16) ? 1 : 0;
  const v8bf* Qv = (const v8bf*)Qs;
  const v8bf* Kv = (const v8bf*)Ks;
  const v8bf* Vv = (const v8bf*)Vt;

  // ---- Phase 1: S[16,64] per wave ----
  v8f sacc[4] = {};
  const int qrow = wave * 16 + (lane & 15);
#pragma unroll
  for (int k0 = 0; k0 < DHEAD; k0 += 32) {
    const int k08 = k0 >> 3;
    v16bf a = frag(Qv[qrow * (QSTR / 8) + k08 + kb8], Qv[qrow * (QSTR / 8) + k08 + kb8 + 2]);
#pragma unroll
    for (int nt = 0; nt < 4; ++nt) {
      const int col = nt * 16 + (lane & 15);
      v16bf b = frag(Kv[col * (QSTR / 8) + k08 + kb8], Kv[col * (QSTR / 8) + k08 + kb8 + 2]);
      sacc[nt] = __builtin_amdgcn_wmma_f32_16x16x32_bf16(false, a, false, b, (short)0, sacc[nt], false, false);
    }
  }
  // ---- Dump S to wave-private LDS region ----
  float* Sw = Ss + wave * 16 * 64;
#pragma unroll
  for (int r = 0; r < 8; ++r) {
    const int sr = r + ((lane & 16) ? 8 : 0);
#pragma unroll
    for (int nt = 0; nt < 4; ++nt)
      Sw[sr * 64 + nt * 16 + (lane & 15)] = sacc[nt][r];
  }
  // ---- Softmax: lanes 0..15 each own one row; masked cols >= 50 ----
  __bf16* Pw = Ps + wave * 16 * QSTR;
  if (lane < 16) {
    float m = -1e30f;
    for (int c = 0; c < LCHUNK; ++c) m = fmaxf(m, Sw[lane * 64 + c]);
    float s = 0.f;
    for (int c = 0; c < LCHUNK; ++c) s += __expf(Sw[lane * 64 + c] - m);
    const float inv = 1.0f / s;
    v8bf* Pp = (v8bf*)Ps + (wave * 16 + lane) * (QSTR / 8);
#pragma unroll
    for (int j = 0; j < 8; ++j) {
      v8bf p;
#pragma unroll
      for (int e = 0; e < 8; ++e) {
        const int c = j * 8 + e;
        p[e] = (c < LCHUNK) ? (__bf16)(__expf(Sw[lane * 64 + c] - m) * inv) : (__bf16)0.0f;
      }
      Pp[j] = p;
    }
  }
  // ---- Phase 2: O[16,64] = P[16,64] @ V[64,64] ----
  const v8bf* Pv = (const v8bf*)Pw;
  v8f oacc[4] = {};
#pragma unroll
  for (int k0 = 0; k0 < 64; k0 += 32) {
    const int k08 = k0 >> 3;
    v16bf a = frag(Pv[(lane & 15) * (QSTR / 8) + k08 + kb8],
                   Pv[(lane & 15) * (QSTR / 8) + k08 + kb8 + 2]);
#pragma unroll
    for (int nt = 0; nt < 4; ++nt) {
      const int col = nt * 16 + (lane & 15);
      v16bf b = frag(Vv[col * (QSTR / 8) + k08 + kb8], Vv[col * (QSTR / 8) + k08 + kb8 + 2]);
      oacc[nt] = __builtin_amdgcn_wmma_f32_16x16x32_bf16(false, a, false, b, (short)0, oacc[nt], false, false);
    }
  }
#pragma unroll
  for (int r = 0; r < 8; ++r) {
    const int row = wave * 16 + r + ((lane & 16) ? 8 : 0);
    if (row < LCHUNK) {
#pragma unroll
      for (int nt = 0; nt < 4; ++nt) {
        const int d = nt * 16 + (lane & 15);
        o[(long)(n * LCHUNK + row) * DMODEL + hd * DHEAD + d] = oacc[nt][r];
      }
    }
  }
}

// =====================================================================
// LayerNorm: one block per row, 256 threads.
// =====================================================================
__global__ __launch_bounds__(256)
void ln_kernel(const float* __restrict__ x, const float* __restrict__ s,
               const float* __restrict__ b, float* __restrict__ out, int D) {
  const int row = blockIdx.x;
  const int tid = threadIdx.x;
  __shared__ float r1[256], r2[256];
  float sum = 0.f, sq = 0.f;
  for (int c = tid; c < D; c += 256) {
    float v = x[(long)row * D + c];
    sum += v; sq += v * v;
  }
  r1[tid] = sum; r2[tid] = sq;
  __syncthreads();
  for (int off = 128; off > 0; off >>= 1) {
    if (tid < off) { r1[tid] += r1[tid + off]; r2[tid] += r2[tid + off]; }
    __syncthreads();
  }
  float mu  = r1[0] / (float)D;
  float var = r2[0] / (float)D - mu * mu;
  float inv = rsqrtf(var + LN_EPS);
  for (int c = tid; c < D; c += 256)
    out[(long)row * D + c] = (x[(long)row * D + c] - mu) * inv * s[c] + b[c];
}

// ---------------- conv weight repack + im2col + pool ----------------
__global__ void repack_conv1_kernel(const float* __restrict__ w, float* __restrict__ B1) {
  int idx = blockIdx.x * 256 + threadIdx.x;
  if (idx >= K1 * DMODEL) return;
  int kk = idx / DMODEL, d = idx % DMODEL;
  int tap = kk / C_MEL, c = kk % C_MEL;
  B1[idx] = w[((long)d * C_MEL + c) * 3 + tap];
}
__global__ void repack_conv2_kernel(const float* __restrict__ w, float* __restrict__ B2) {
  int idx = blockIdx.x * 256 + threadIdx.x;
  if (idx >= K2 * DMODEL) return;
  int kk = idx / DMODEL, d = idx % DMODEL;
  int tap = kk / DMODEL, c = kk % DMODEL;
  B2[idx] = w[((long)d * DMODEL + c) * 3 + tap];
}
__global__ void im2col1_kernel(const float* __restrict__ x, float* __restrict__ A1) {
  int idx = blockIdx.x * 256 + threadIdx.x;
  if (idx >= MCONV1 * K1) return;
  int m = idx / K1, kk = idx % K1;
  int n = m / BASE, t = m % BASE;
  int tap = kk / C_MEL, c = kk % C_MEL;
  int tt = t - 1 + tap;
  A1[idx] = (tt >= 0 && tt < BASE) ? x[(long)c * T_IN + n * BASE + tt] : 0.0f;
}
__global__ void im2col2_kernel(const float* __restrict__ x1, float* __restrict__ A2) {
  int idx = blockIdx.x * 256 + threadIdx.x;
  if (idx >= NTOK * K2) return;
  int m = idx / K2, kk = idx % K2;
  int n = m / LCHUNK, l = m % LCHUNK;
  int tap = kk / DMODEL, c = kk % DMODEL;
  int tt = 2 * l - 1 + tap;
  A2[idx] = (tt >= 0 && tt < BASE) ? x1[(long)(n * BASE + tt) * DMODEL + c] : 0.0f;
}
__global__ void pool_kernel(const float* __restrict__ h, float* __restrict__ p) {
  int idx = blockIdx.x * 256 + threadIdx.x;
  if (idx >= 750 * DMODEL) return;
  int m = idx / DMODEL, d = idx % DMODEL;
  int n = m / 25, i = m % 25;
  float a = h[(long)(n * LCHUNK + 2 * i) * DMODEL + d];
  float b = h[(long)(n * LCHUNK + 2 * i + 1) * DMODEL + d];
  p[idx] = 0.5f * (a + b);
}

// ---------------- host-side helpers ----------------
static inline void launch_gemm(const float* A, const float* B, const float* bias,
                               const float* res, int res_mod, float scale, int gelu,
                               float* out, int M, int N, int K, hipStream_t s) {
  dim3 grid(N / 64, (M + 127) / 128);
  gemm_wmma_bf16<<<grid, 256, 0, s>>>(A, B, bias, res, res_mod, scale, gelu, out, M, N, K);
}

extern "C" void kernel_launch(void* const* d_in, const int* in_sizes, int n_in,
                              void* d_out, int out_size, void* d_ws, size_t ws_size,
                              hipStream_t stream) {
  (void)in_sizes; (void)n_in; (void)out_size; (void)ws_size;
  const float* input_features = (const float*)d_in[0];
  const float* conv1_w = (const float*)d_in[1];
  const float* conv1_b = (const float*)d_in[2];
  const float* conv2_w = (const float*)d_in[3];
  const float* conv2_b = (const float*)d_in[4];
  const float* pos_emb = (const float*)d_in[5];
  const float* ln1_s = (const float*)d_in[6];
  const float* ln1_b = (const float*)d_in[7];
  const float* wq = (const float*)d_in[8];
  const float* bq = (const float*)d_in[9];
  const float* wk = (const float*)d_in[10];
  const float* bk = (const float*)d_in[11];
  const float* wv = (const float*)d_in[12];
  const float* bv = (const float*)d_in[13];
  const float* wo = (const float*)d_in[14];
  const float* bo = (const float*)d_in[15];
  const float* ln2_s = (const float*)d_in[16];
  const float* ln2_b = (const float*)d_in[17];
  const float* fc1_w = (const float*)d_in[18];
  const float* fc1_b = (const float*)d_in[19];
  const float* fc2_w = (const float*)d_in[20];
  const float* fc2_b = (const float*)d_in[21];
  const float* lnp_s = (const float*)d_in[22];
  const float* lnp_b = (const float*)d_in[23];
  const float* proj_w = (const float*)d_in[24];
  const float* proj_b = (const float*)d_in[25];
  float* out = (float*)d_out;

  float* ws = (float*)d_ws;
  size_t off = 0;
  float* A1   = ws + off; off += (size_t)MCONV1 * K1;
  float* x1   = ws + off; off += (size_t)MCONV1 * DMODEL;
  float* A2   = ws + off; off += (size_t)NTOK * K2;
  float* hbuf = ws + off; off += (size_t)NTOK * DMODEL;
  float* rbuf = ws + off; off += (size_t)NTOK * DMODEL;
  float* qb   = ws + off; off += (size_t)NTOK * DMODEL;
  float* kb   = ws + off; off += (size_t)NTOK * DMODEL;
  float* vb   = ws + off; off += (size_t)NTOK * DMODEL;
  float* ob   = ws + off; off += (size_t)NTOK * DMODEL;
  float* tff  = ws + off; off += (size_t)NTOK * DFF;
  float* pooled = ws + off; off += (size_t)750 * DMODEL;
  float* lnp  = ws + off; off += (size_t)750 * DMODEL;
  float* B1   = ws + off; off += (size_t)K1 * DMODEL;
  float* B2   = ws + off; off += (size_t)K2 * DMODEL;

  const float qscale = 0.125f;  // DH^-0.5

  repack_conv1_kernel<<<(K1 * DMODEL + 255) / 256, 256, 0, stream>>>(conv1_w, B1);
  repack_conv2_kernel<<<(K2 * DMODEL + 255) / 256, 256, 0, stream>>>(conv2_w, B2);

  im2col1_kernel<<<(MCONV1 * K1 + 255) / 256, 256, 0, stream>>>(input_features, A1);
  launch_gemm(A1, B1, conv1_b, nullptr, 0, 1.0f, 1, x1, MCONV1, DMODEL, K1, stream);

  im2col2_kernel<<<(NTOK * K2 + 255) / 256, 256, 0, stream>>>(x1, A2);
  launch_gemm(A2, B2, conv2_b, pos_emb, LCHUNK, 1.0f, 1, hbuf, NTOK, DMODEL, K2, stream);

  for (int l = 0; l < NLAYER; ++l) {
    const long dd = (long)l * DMODEL * DMODEL;
    ln_kernel<<<NTOK, 256, 0, stream>>>(hbuf, ln1_s + l * DMODEL, ln1_b + l * DMODEL, rbuf, DMODEL);
    launch_gemm(rbuf, wq + dd, bq + l * DMODEL, nullptr, 0, qscale, 0, qb, NTOK, DMODEL, DMODEL, stream);
    launch_gemm(rbuf, wk + dd, bk + l * DMODEL, nullptr, 0, 1.0f, 0, kb, NTOK, DMODEL, DMODEL, stream);
    launch_gemm(rbuf, wv + dd, bv + l * DMODEL, nullptr, 0, 1.0f, 0, vb, NTOK, DMODEL, DMODEL, stream);
    attn_wmma_kernel<<<NCHUNK * NHEAD, 128, 0, stream>>>(qb, kb, vb, ob);
    launch_gemm(ob, wo + dd, bo + l * DMODEL, hbuf, 0, 1.0f, 0, hbuf, NTOK, DMODEL, DMODEL, stream);
    ln_kernel<<<NTOK, 256, 0, stream>>>(hbuf, ln2_s + l * DMODEL, ln2_b + l * DMODEL, rbuf, DMODEL);
    launch_gemm(rbuf, fc1_w + (long)l * DMODEL * DFF, fc1_b + l * DFF, nullptr, 0, 1.0f, 1,
                tff, NTOK, DFF, DMODEL, stream);
    launch_gemm(tff, fc2_w + (long)l * DFF * DMODEL, fc2_b + l * DMODEL, hbuf, 0, 1.0f, 0,
                hbuf, NTOK, DMODEL, DFF, stream);
  }

  pool_kernel<<<(750 * DMODEL + 255) / 256, 256, 0, stream>>>(hbuf, pooled);
  ln_kernel<<<750, 256, 0, stream>>>(pooled, lnp_s, lnp_b, lnp, DMODEL);
  launch_gemm(lnp, proj_w, proj_b, nullptr, 0, 1.0f, 0, out, 750, DOUTP, DMODEL, stream);
}
